// LocalPool_VN_Pointnet_57784490001123
// MI455X (gfx1250) — matrix operand order repaired and checked by reading, not compiled, not run on hardware
//
#include <hip/hip_runtime.h>

#define NPB   4096          // points per batch
#define BATCH 4
#define NPTS  16384         // BATCH*NPB
#define NCOL  49152         // 3*NPTS
#define HIDC  128
#define KNNK  10
#define EPSF  1e-6f

typedef __attribute__((ext_vector_type(16))) _Float16 v16h;
typedef __attribute__((ext_vector_type(8)))  float    v8f;

// ---------------------------------------------------------------------------
// Weight pre-conversion f32 -> f16 (runs once per launch, amortized over the
// 768-block GEMM grids that reuse every matrix).
// ---------------------------------------------------------------------------
__global__ __launch_bounds__(256)
void cvt_f16_kernel(const float* __restrict__ src, _Float16* __restrict__ dst, int n)
{
    const int idx = blockIdx.x * blockDim.x + threadIdx.x;
    if (idx < n) dst[idx] = (_Float16)src[idx];
}

// ---------------------------------------------------------------------------
// Kernel 1: kNN (top-10 by squared distance, includes self) + edge features
// (e = q-p, x = p, c = q x p) + first VN layer (Wcp_f/Wcp_d, slope 0) + mean
// over K.  One wave per point; batch point cloud cached in LDS.
// Output layout: out[ch*NCOL + d*NPTS + pt], ch<128, d<3.
// ---------------------------------------------------------------------------
__global__ __launch_bounds__(128)
void knn_vn_kernel(const float* __restrict__ p,
                   const float* __restrict__ Wf,
                   const float* __restrict__ Wd,
                   float* __restrict__ out)
{
    __shared__ float lp[NPB * 3];
    const int tid  = threadIdx.x;
    const int wave = tid >> 5;
    const int lane = tid & 31;
    const int ptBase = blockIdx.x * 4;          // 4 points (waves) per block
    const int b = ptBase >> 12;                 // ptBase / NPB

    const float* pb = p + b * NPB * 3;
    for (int i = tid; i < NPB * 3; i += blockDim.x) lp[i] = pb[i];
    __syncthreads();

    const int pt = ptBase + wave;
    const int n  = pt & (NPB - 1);
    const float px = lp[n*3+0], py = lp[n*3+1], pz = lp[n*3+2];

    // per-lane sorted top-10 (smallest d^2) over 128 candidates
    float bd[KNNK]; int bi[KNNK];
    #pragma unroll
    for (int i = 0; i < KNNK; ++i) { bd[i] = 3.4e38f; bi[i] = 0; }
    for (int j = 0; j < NPB / 32; ++j) {
        const int m = lane + (j << 5);
        const float dx = lp[m*3+0] - px;
        const float dy = lp[m*3+1] - py;
        const float dz = lp[m*3+2] - pz;
        const float d2 = dx*dx + dy*dy + dz*dz;
        if (d2 < bd[KNNK-1]) {
            bd[KNNK-1] = d2; bi[KNNK-1] = m;
            #pragma unroll
            for (int s = KNNK-1; s >= 1; --s) {
                if (bd[s] < bd[s-1]) {
                    float tf = bd[s]; bd[s] = bd[s-1]; bd[s-1] = tf;
                    int   ti = bi[s]; bi[s] = bi[s-1]; bi[s-1] = ti;
                }
            }
        }
    }

    // 10-round wave-wide merge of the 32 sorted lists (heads at index 0)
    int nb[KNNK];
    #pragma unroll
    for (int t = 0; t < KNNK; ++t) {
        float v = bd[0]; int who = lane;
        #pragma unroll
        for (int off = 16; off >= 1; off >>= 1) {
            const float v2 = __shfl_xor(v, off, 32);
            const int   w2 = __shfl_xor(who, off, 32);
            if (v2 < v || (v2 == v && w2 < who)) { v = v2; who = w2; }
        }
        nb[t] = __shfl(bi[0], who, 32);
        if (lane == who) {
            #pragma unroll
            for (int s = 0; s < KNNK-1; ++s) { bd[s] = bd[s+1]; bi[s] = bi[s+1]; }
            bd[KNNK-1] = 3.4e38f;
        }
    }

    // fused first VN layer: channels ch = lane + 32*c
    float wf[4][3], wd[4][3], acc[4][3];
    #pragma unroll
    for (int c = 0; c < 4; ++c) {
        const int ch = lane + (c << 5);
        #pragma unroll
        for (int i = 0; i < 3; ++i) { wf[c][i] = Wf[ch*3+i]; wd[c][i] = Wd[ch*3+i]; }
        acc[c][0] = acc[c][1] = acc[c][2] = 0.f;
    }
    #pragma unroll
    for (int t = 0; t < KNNK; ++t) {
        const int m = nb[t];
        const float qx = lp[m*3+0], qy = lp[m*3+1], qz = lp[m*3+2];
        const float e0 = qx - px, e1 = qy - py, e2 = qz - pz;
        const float c0 = qy*pz - qz*py;     // cross(neigh, x)
        const float c1 = qz*px - qx*pz;
        const float c2 = qx*py - qy*px;
        #pragma unroll
        for (int c = 0; c < 4; ++c) {
            float f0 = wf[c][0]*e0 + wf[c][1]*px + wf[c][2]*c0;
            float f1 = wf[c][0]*e1 + wf[c][1]*py + wf[c][2]*c1;
            float f2 = wf[c][0]*e2 + wf[c][1]*pz + wf[c][2]*c2;
            const float d0 = wd[c][0]*e0 + wd[c][1]*px + wd[c][2]*c0;
            const float d1 = wd[c][0]*e1 + wd[c][1]*py + wd[c][2]*c1;
            const float d2 = wd[c][0]*e2 + wd[c][1]*pz + wd[c][2]*c2;
            const float dot = f0*d0 + f1*d1 + f2*d2;
            const float dsq = d0*d0 + d1*d1 + d2*d2;
            if (dot < 0.f) {
                const float s = dot / (dsq + EPSF);
                f0 -= s*d0; f1 -= s*d1; f2 -= s*d2;
            }
            acc[c][0] += f0; acc[c][1] += f1; acc[c][2] += f2;
        }
    }
    const float inv = 1.0f / (float)KNNK;
    #pragma unroll
    for (int c = 0; c < 4; ++c) {
        const int ch = lane + (c << 5);
        #pragma unroll
        for (int d = 0; d < 3; ++d)
            out[ch*NCOL + d*NPTS + pt] = acc[c][d] * inv;
    }
}

// ---------------------------------------------------------------------------
// WMMA GEMM: Y[M x NCOL] = Wh[M x K] * X[K x NCOL] (+ optional residual R).
// Wh is pre-converted f16.  Block = (M/16) waves, 64 columns per block.
// X tile staged as f16 in LDS (padded stride 136 halves).
// ---------------------------------------------------------------------------
template <bool HASR>
__global__ __launch_bounds__(256)
void gemm_vn(const _Float16* __restrict__ Wh,
             const float* __restrict__ X,
             float* __restrict__ Y,
             const float* __restrict__ R,
             int M, int K)
{
    __shared__ _Float16 ldsX[64 * 136];
    const int tid  = threadIdx.x;
    const int col0 = blockIdx.x * 64;

    for (int e = tid; e < K * 64; e += blockDim.x) {
        const int k = e >> 6, c = e & 63;
        ldsX[c * 136 + k] = (_Float16)X[k * NCOL + col0 + c];
    }
    __syncthreads();

    const int wave = tid >> 5, lane = tid & 31;
    const int m0 = wave << 4;
    const int ln = lane & 15;
    const int hi = lane >> 4;

    v8f acc[4] = {};
    for (int k0 = 0; k0 < K; k0 += 32) {
        // A fragment: lane<16 holds K {0..15}, lane>=16 holds {16..31}
        v16h afrag;
        const _Float16* wr = Wh + (m0 + ln) * K + k0 + (hi << 4);
        #pragma unroll
        for (int i = 0; i < 8; ++i) afrag[i]     = wr[i];
        #pragma unroll
        for (int i = 0; i < 8; ++i) afrag[8 + i] = wr[8 + i];

        #pragma unroll
        for (int s = 0; s < 4; ++s) {
            // B fragment: lanes 0-15 -> K=k0..k0+15, lanes 16-31 -> K=k0+16..k0+31
            v16h bfrag;
            const _Float16* xr = &ldsX[(s*16 + ln) * 136 + k0 + (hi << 4)];
            #pragma unroll
            for (int i = 0; i < 16; ++i) bfrag[i] = xr[i];
            acc[s] = __builtin_amdgcn_wmma_f32_16x16x32_f16(
                false, afrag, false, bfrag, (short)0, acc[s], false, false);
        }
    }

    const int mrow = m0 + (hi ? 8 : 0);
    #pragma unroll
    for (int s = 0; s < 4; ++s) {
        const int col = col0 + s*16 + ln;
        #pragma unroll
        for (int r = 0; r < 8; ++r) {
            const int idx = (mrow + r) * NCOL + col;
            float v = acc[s][r];
            if constexpr (HASR) v += R[idx];
            Y[idx] = v;
        }
    }
}

// ---------------------------------------------------------------------------
// VN activation: T = slope*F + (1-slope)*(dot>=0 ? F : F - dot/(|D|^2+eps)*D)
// over the 3 vector components per (ch, pt).  Safe for T == D (in place).
// ---------------------------------------------------------------------------
__global__ __launch_bounds__(256)
void vn_act_kernel(const float* __restrict__ F,
                   const float* __restrict__ D,
                   float* __restrict__ T,
                   float slope, int chN)
{
    const int idx = blockIdx.x * blockDim.x + threadIdx.x;
    if (idx >= chN * NPTS) return;
    const int pt = idx & (NPTS - 1);
    const int ch = idx >> 14;
    const int base = ch * NCOL + pt;
    const float f0 = F[base], f1 = F[base + NPTS], f2 = F[base + 2*NPTS];
    const float d0 = D[base], d1 = D[base + NPTS], d2 = D[base + 2*NPTS];
    const float dot = f0*d0 + f1*d1 + f2*d2;
    const float dsq = d0*d0 + d1*d1 + d2*d2;
    const float s   = dot / (dsq + EPSF);
    const bool  neg = (dot < 0.f);
    const float a0 = neg ? f0 - s*d0 : f0;
    const float a1 = neg ? f1 - s*d1 : f1;
    const float a2 = neg ? f2 - s*d2 : f2;
    const float omk = 1.0f - slope;
    T[base]          = slope*f0 + omk*a0;
    T[base + NPTS]   = slope*f1 + omk*a1;
    T[base + 2*NPTS] = slope*f2 + omk*a2;
}

// ---------------------------------------------------------------------------
// Frame head: z0 = Wslin(2x32) * Z, Gram-Schmidt -> orthonormal frame
// columns [u1 u2 u3]; frames[pt*9 + j*3 + k] = u_k[j].
// ---------------------------------------------------------------------------
__global__ __launch_bounds__(128)
void frame_kernel(const float* __restrict__ Z,
                  const float* __restrict__ Wlin,
                  float* __restrict__ frames)
{
    const int pt = blockIdx.x * blockDim.x + threadIdx.x;
    if (pt >= NPTS) return;
    float v1[3] = {0,0,0}, v2[3] = {0,0,0};
    for (int i = 0; i < 32; ++i) {
        const float w0 = Wlin[i], w1 = Wlin[32 + i];
        #pragma unroll
        for (int d = 0; d < 3; ++d) {
            const float z = Z[i*NCOL + d*NPTS + pt];
            v1[d] += w0 * z; v2[d] += w1 * z;
        }
    }
    const float n1 = sqrtf(v1[0]*v1[0] + v1[1]*v1[1] + v1[2]*v1[2]) + EPSF;
    float u1[3] = { v1[0]/n1, v1[1]/n1, v1[2]/n1 };
    const float dp = v2[0]*u1[0] + v2[1]*u1[1] + v2[2]*u1[2];
    #pragma unroll
    for (int d = 0; d < 3; ++d) v2[d] -= dp * u1[d];
    const float n2 = sqrtf(v2[0]*v2[0] + v2[1]*v2[1] + v2[2]*v2[2]) + EPSF;
    float u2[3] = { v2[0]/n2, v2[1]/n2, v2[2]/n2 };
    float u3[3] = { u1[1]*u2[2] - u1[2]*u2[1],
                    u1[2]*u2[0] - u1[0]*u2[2],
                    u1[0]*u2[1] - u1[1]*u2[0] };
    float* fr = frames + pt * 9;
    #pragma unroll
    for (int j = 0; j < 3; ++j) {
        fr[j*3 + 0] = u1[j]; fr[j*3 + 1] = u2[j]; fr[j*3 + 2] = u3[j];
    }
}

// x_inv: c[pt*128+ch] = sum_k ( sum_j x[ch][j][pt] * frame[j][k] )^2
__global__ __launch_bounds__(256)
void cinv_kernel(const float* __restrict__ A,
                 const float* __restrict__ frames,
                 float* __restrict__ c)
{
    const int idx = blockIdx.x * blockDim.x + threadIdx.x;
    if (idx >= HIDC * NPTS) return;
    const int pt = idx & (NPTS - 1);
    const int ch = idx >> 14;
    const float x0 = A[ch*NCOL + pt];
    const float x1 = A[ch*NCOL + NPTS + pt];
    const float x2 = A[ch*NCOL + 2*NPTS + pt];
    const float* fr = frames + pt * 9;
    float s = 0.f;
    #pragma unroll
    for (int k = 0; k < 3; ++k) {
        const float t = x0*fr[k] + x1*fr[3+k] + x2*fr[6+k];
        s += t * t;
    }
    c[pt * HIDC + ch] = s;
}

__global__ __launch_bounds__(256)
void zero_kernel(float* __restrict__ ptr, int n)
{
    const int idx = blockIdx.x * blockDim.x + threadIdx.x;
    if (idx < n) ptr[idx] = 0.f;
}

// scatter-mean accumulation: one wave per (plane, point); lanes cover channels
__global__ __launch_bounds__(256)
void scatter_kernel(const float* __restrict__ p,
                    const float* __restrict__ c,
                    float* __restrict__ sums,
                    float* __restrict__ cnts)
{
    const int gw   = (blockIdx.x * blockDim.x + threadIdx.x) >> 5;
    const int lane = threadIdx.x & 31;
    if (gw >= 3 * NPTS) return;
    const int plane = gw >> 14;
    const int pt    = gw & (NPTS - 1);
    const int b = pt >> 12, n = pt & (NPB - 1);
    const float* pp = p + (b * NPB + n) * 3;
    const float pv[3] = { pp[0], pp[1], pp[2] };
    int a0, a1;
    if (plane == 0)      { a0 = 0; a1 = 2; }   // fxz
    else if (plane == 1) { a0 = 0; a1 = 1; }   // fxy
    else                 { a0 = 1; a1 = 2; }   // fyz
    float x = pv[a0] / 1.101f + 0.5f;
    float y = pv[a1] / 1.101f + 0.5f;
    x = (x >= 1.f) ? (1.f - 1e-5f) : (x < 0.f ? 0.f : x);
    y = (y >= 1.f) ? (1.f - 1e-5f) : (y < 0.f ? 0.f : y);
    const int cell = (int)(x * 64.f) + 64 * (int)(y * 64.f);
    const int base = ((plane*BATCH + b) * 4096 + cell) * HIDC;
    #pragma unroll
    for (int q = 0; q < 4; ++q) {
        const int ch = lane + (q << 5);
        atomicAdd(&sums[base + ch], c[pt * HIDC + ch]);
    }
    if (lane == 0) atomicAdd(&cnts[(plane*BATCH + b) * 4096 + cell], 1.0f);
}

// normalize + transpose [plane][b][cell][ch] -> out[plane][b][ch][cell]
__global__ __launch_bounds__(256)
void finalize_kernel(const float* __restrict__ sums,
                     const float* __restrict__ cnts,
                     float* __restrict__ out)
{
    const int idx = blockIdx.x * blockDim.x + threadIdx.x;
    if (idx >= 3 * BATCH * HIDC * 4096) return;
    const int cell = idx & 4095;
    const int ch   = (idx >> 12) & (HIDC - 1);
    const int pb   = idx >> 19;
    const float cnt = cnts[pb * 4096 + cell];
    out[idx] = sums[(pb * 4096 + cell) * HIDC + ch] / fmaxf(cnt, 1.0f);
}

// ---------------------------------------------------------------------------
extern "C" void kernel_launch(void* const* d_in, const int* in_sizes, int n_in,
                              void* d_out, int out_size, void* d_ws, size_t ws_size,
                              hipStream_t stream) {
    const float* p     = (const float*)d_in[0];
    const float* Wcp_f = (const float*)d_in[1];
    const float* Wcp_d = (const float*)d_in[2];
    const float* Wfc   = (const float*)d_in[3];
    const float* Wd0s  = (const float*)d_in[4];
    const float* W0s   = (const float*)d_in[5];
    const float* Wd1s  = (const float*)d_in[6];
    const float* W1s   = (const float*)d_in[7];
    const float* Ws1f  = (const float*)d_in[8];
    const float* Ws1d  = (const float*)d_in[9];
    const float* Ws2f  = (const float*)d_in[10];
    const float* Ws2d  = (const float*)d_in[11];
    const float* Wslin = (const float*)d_in[12];

    const size_t TSZ = (size_t)HIDC * NCOL;       // 6,291,456 floats
    float* A      = (float*)d_ws;
    float* B1     = A  + TSZ;
    float* B2     = B1 + TSZ;
    float* B3     = B2 + TSZ;
    float* cbuf   = B3 + TSZ;                     // NPTS*128 floats
    float* frames = cbuf + (size_t)NPTS * HIDC;   // NPTS*9 floats
    float* sums   = B1;                           // reuse (exactly TSZ floats)
    float* cnts   = B2;                           // reuse (49,152 floats)

    // f16 weight arena (after frames)
    _Float16* wh     = (_Float16*)(frames + (size_t)NPTS * 9);
    _Float16* Wfc_h  = wh;                        // 16384
    _Float16* Wd0_h  = Wfc_h + 16384;             // 5*16384
    _Float16* W0_h   = Wd0_h + 5*16384;
    _Float16* Wd1_h  = W0_h  + 5*16384;
    _Float16* W1_h   = Wd1_h + 5*16384;
    _Float16* Ws1f_h = W1_h  + 5*16384;           // 8192
    _Float16* Ws1d_h = Ws1f_h + 8192;
    _Float16* Ws2f_h = Ws1d_h + 8192;             // 2048
    _Float16* Ws2d_h = Ws2f_h + 2048;

    const dim3 g768(768), b256(256), b128(128), b64(64);

    // 0) pre-convert weights to f16
    cvt_f16_kernel<<<64, b256, 0, stream>>>(Wfc,  Wfc_h,  16384);
    cvt_f16_kernel<<<320, b256, 0, stream>>>(Wd0s, Wd0_h, 5*16384);
    cvt_f16_kernel<<<320, b256, 0, stream>>>(W0s,  W0_h,  5*16384);
    cvt_f16_kernel<<<320, b256, 0, stream>>>(Wd1s, Wd1_h, 5*16384);
    cvt_f16_kernel<<<320, b256, 0, stream>>>(W1s,  W1_h,  5*16384);
    cvt_f16_kernel<<<32, b256, 0, stream>>>(Ws1f, Ws1f_h, 8192);
    cvt_f16_kernel<<<32, b256, 0, stream>>>(Ws1d, Ws1d_h, 8192);
    cvt_f16_kernel<<<8,  b256, 0, stream>>>(Ws2f, Ws2f_h, 2048);
    cvt_f16_kernel<<<8,  b256, 0, stream>>>(Ws2d, Ws2d_h, 2048);

    // 1) kNN + edge features + first VN layer + mean over K  -> B3
    knn_vn_kernel<<<NPTS/4, 128, 0, stream>>>(p, Wcp_f, Wcp_d, B3);

    // 2) A = Wfc * B3
    gemm_vn<false><<<g768, b256, 0, stream>>>(Wfc_h, B3, A, nullptr, 128, 128);

    // 3) 5 resblocks
    for (int i = 0; i < 5; ++i) {
        const _Float16* Wd0 = Wd0_h + (size_t)i * 16384;
        const _Float16* W0  = W0_h  + (size_t)i * 16384;
        const _Float16* Wd1 = Wd1_h + (size_t)i * 16384;
        const _Float16* W1  = W1_h  + (size_t)i * 16384;
        gemm_vn<false><<<g768, b256, 0, stream>>>(Wd0, A, B1, nullptr, 128, 128);
        vn_act_kernel<<<(128*NPTS)/256, b256, 0, stream>>>(A, B1, B1, 0.0f, 128);
        gemm_vn<false><<<g768, b256, 0, stream>>>(W0, B1, B2, nullptr, 128, 128);
        gemm_vn<false><<<g768, b256, 0, stream>>>(Wd1, B2, B3, nullptr, 128, 128);
        vn_act_kernel<<<(128*NPTS)/256, b256, 0, stream>>>(B2, B3, B3, 0.0f, 128);
        gemm_vn<true><<<g768, b256, 0, stream>>>(W1, B3, A, /*residual=*/A, 128, 128);
    }

    // 4) std_feature head
    gemm_vn<false><<<g768, b128, 0, stream>>>(Ws1f_h, A, B1, nullptr, 64, 128);
    gemm_vn<false><<<g768, b128, 0, stream>>>(Ws1d_h, A, B2, nullptr, 64, 128);
    vn_act_kernel<<<(64*NPTS)/256, b256, 0, stream>>>(B1, B2, B1, 0.2f, 64);
    gemm_vn<false><<<g768, b64, 0, stream>>>(Ws2f_h, B1, B2, nullptr, 32, 64);
    gemm_vn<false><<<g768, b64, 0, stream>>>(Ws2d_h, B1, B3, nullptr, 32, 64);
    vn_act_kernel<<<(32*NPTS)/256, b256, 0, stream>>>(B2, B3, B2, 0.2f, 32);

    // 5) frame + rotation-invariant features
    frame_kernel<<<NPTS/128, b128, 0, stream>>>(B2, Wslin, frames);
    cinv_kernel<<<(128*NPTS)/256, b256, 0, stream>>>(A, frames, cbuf);

    // 6) plane features (scatter-mean on 3 planes)
    zero_kernel<<<(int)((TSZ + 255)/256), b256, 0, stream>>>(sums, (int)TSZ);
    zero_kernel<<<(3*BATCH*4096 + 255)/256, b256, 0, stream>>>(cnts, 3*BATCH*4096);
    scatter_kernel<<<(3*NPTS)/8, b256, 0, stream>>>(p, cbuf, sums, cnts);
    finalize_kernel<<<(3*BATCH*HIDC*4096)/256, b256, 0, stream>>>(sums, cnts, (float*)d_out);
}
